// TimeSFormerXAttention_695784702465
// MI455X (gfx1250) — matrix operand order repaired and checked by reading, not compiled, not run on hardware
//
#include <hip/hip_runtime.h>
#include <hip/hip_bf16.h>

// ---------------------------------------------------------------------------
// TimeSFormer cross-attention for MI455X (gfx1250), bf16 WMMA path.
// B=4, LV=16, HW=196, LT=32, D=768, H=12, DH=64. Keys padded: vid 0..195,
// pad 196..207, txt 208..239, pad 240..255  -> LKP=256 (multiple of 32).
// ---------------------------------------------------------------------------

#define B_    4
#define LV_   16
#define HW_   196
#define LT_   32
#define D_    768
#define H_    12
#define DH_   64
#define LKP   256
#define SCALE_ 0.125f   // 1/sqrt(64)

typedef __attribute__((ext_vector_type(16))) __bf16 v16bf;
typedef __attribute__((ext_vector_type(8)))  __bf16 v8bf;
typedef __attribute__((ext_vector_type(8)))  float  v8f;

__device__ __forceinline__ __bf16 f2bf(float f) {
    unsigned u = __builtin_bit_cast(unsigned, f);
    unsigned r = u + 0x7FFFu + ((u >> 16) & 1u);   // round-to-nearest-even
    unsigned short h = (unsigned short)(r >> 16);
    return __builtin_bit_cast(__bf16, h);
}

// ---------------------------------------------------------------------------
// f32 -> bf16 conversion
// ---------------------------------------------------------------------------
__global__ __launch_bounds__(256) void cvt_kernel(const float* __restrict__ in,
                                                  __bf16* __restrict__ out,
                                                  long long n) {
    long long i = (long long)blockIdx.x * 256 + threadIdx.x;
    long long stride = (long long)gridDim.x * 256;
    for (; i < n; i += stride) out[i] = f2bf(in[i]);
}

// ---------------------------------------------------------------------------
// Projection GEMM: y[t][o] = sum_i x[t][i] * W[o][i] + b[o]
// 8 waves: wm in {0,1} (M=32 tile), wn in {0..3} (N=256 tile), K-loop 768/32.
// ---------------------------------------------------------------------------
struct ProjDesc {
    const __bf16* x;     // [T][768] bf16 tokens
    const __bf16* w;     // [768][768] bf16, row-major [out][in]
    const float*  bias;  // [768] f32
    __bf16*       out;   // normal: [T][768]; transposed: [bv][768][LKP]
    int T;               // token count (12544 or 2048)
    int Ltok;            // tokens per (b,v): 196 or 32
    int transposed;      // 1 for V projections
    int key_off;         // 0 (video part) or 208 (text part)
};
struct ProjArgs { ProjDesc d[12]; };

__global__ __launch_bounds__(256) void proj_kernel(ProjArgs args) {
    ProjDesc d = args.d[blockIdx.z];
    int m0 = blockIdx.x * 32;
    if (m0 >= d.T) return;
    int n0 = blockIdx.y * 256;

    int tid  = threadIdx.x;
    int lane = tid & 31;
    int wave = tid >> 5;
    int wm = wave >> 2;           // 0..1
    int wn = wave & 3;            // 0..3
    int mbase = m0 + wm * 16;
    int nbase = n0 + wn * 64;

    int alo = (lane & 16) ? 8 : 0;        // A: K sub-offset per half-wave
    int bhi = (lane & 16) ? 8 : 0;        // B: K sub-offset per half-wave
    int arow = mbase + (lane & 15);

    v8f acc[4];
    #pragma unroll
    for (int j = 0; j < 4; ++j) acc[j] = (v8f){};

    const __bf16* xrow = d.x + (size_t)arow * D_;
    for (int k = 0; k < D_; k += 32) {
        v16bf a;
        {
            const __bf16* p = xrow + k + alo;
            ((v8bf*)&a)[0] = *(const v8bf*)(p);
            ((v8bf*)&a)[1] = *(const v8bf*)(p + 16);
        }
        int bk = k + bhi;
        #pragma unroll
        for (int j = 0; j < 4; ++j) {
            const __bf16* q = d.w + (size_t)(nbase + j * 16 + (lane & 15)) * D_ + bk;
            v16bf bb;
            ((v8bf*)&bb)[0] = *(const v8bf*)(q);
            ((v8bf*)&bb)[1] = *(const v8bf*)(q + 16);
            acc[j] = __builtin_amdgcn_wmma_f32_16x16x32_bf16(
                false, a, false, bb, (short)0, acc[j], false, false);
        }
    }

    #pragma unroll
    for (int j = 0; j < 4; ++j) {
        int ocol = nbase + j * 16 + (lane & 15);
        float bv = d.bias[ocol];
        #pragma unroll
        for (int r = 0; r < 8; ++r) {
            int mrow = mbase + r + ((lane & 16) ? 8 : 0);
            float val = acc[j][r] + bv;
            if (!d.transposed) {
                d.out[(size_t)mrow * D_ + ocol] = f2bf(val);
            } else {
                int bvq = mrow / d.Ltok;
                int loc = mrow - bvq * d.Ltok;
                d.out[((size_t)(bvq * D_ + ocol)) * LKP + d.key_off + loc] = f2bf(val);
            }
        }
    }
}

// ---------------------------------------------------------------------------
// Attention: one block (128 thr, 4 waves) per (bv, head, 16-query tile).
// Phase 1: S = Q K^T (per key tile: video tiles use Qa/Kvid, text use Qb/Ktxt)
// Phase 2: masked softmax in LDS, scaled by 1/sqrt(DH) AFTER softmax
// Phase 3: O = P V^T  (V stored transposed [bv][o][key])
// ---------------------------------------------------------------------------
__global__ __launch_bounds__(128) void attn_kernel(
    const __bf16* __restrict__ Qa,    // queries vs video keys   [bv*Lq][768]
    const __bf16* __restrict__ Qb,    // queries vs text keys    [bv*Lq][768]
    const __bf16* __restrict__ Kvid,  // video keys              [bv*196][768]
    const __bf16* __restrict__ Ktxt,  // text keys               [bv*32][768]
    const __bf16* __restrict__ VT,    // values transposed       [bv][768][LKP]
    const unsigned char* __restrict__ tmask,  // [B][32]
    float* __restrict__ out,          // [bv*Lq][768]
    int Lq, int textq)
{
    __shared__ float  S[16 * LKP];
    __shared__ __bf16 P[16 * LKP];
    __shared__ float  red[16 * 8];
    __shared__ float  rowstat[16];

    int qt = blockIdx.x, h = blockIdx.y, bv = blockIdx.z;
    int b = bv >> 4;
    int tid = threadIdx.x, lane = tid & 31, wave = tid >> 5;
    int q0 = qt * 16;
    int alo = (lane & 16) ? 8 : 0;
    int bhi = (lane & 16) ? 8 : 0;
    int arow = q0 + (lane & 15);
    int arow_c = arow < Lq ? arow : Lq - 1;   // clamp pad query rows

    // ---- Phase 1: logits ----
    for (int nt = wave; nt < 16; nt += 4) {
        int col0 = nt * 16;
        v8f acc = (v8f){};
        bool isPad = (nt == 15);
        bool isVid = (nt < 13);
        if (!isPad) {
            const __bf16* Qp = isVid ? Qa : Qb;
            const __bf16* Kp = isVid ? Kvid : Ktxt;
            int Lk = isVid ? HW_ : LT_;
            int kb = isVid ? 0 : 208;
            int kr = col0 - kb + (lane & 15);
            int krc = kr < Lk ? kr : Lk - 1;   // clamp pad key rows
            const __bf16* qbase = Qp + (size_t)(bv * Lq + arow_c) * D_ + h * DH_;
            const __bf16* kbase = Kp + (size_t)(bv * Lk + krc) * D_ + h * DH_;
            #pragma unroll
            for (int kk = 0; kk < DH_; kk += 32) {
                v16bf a, bb;
                const __bf16* pa = qbase + kk + alo;
                ((v8bf*)&a)[0] = *(const v8bf*)(pa);
                ((v8bf*)&a)[1] = *(const v8bf*)(pa + 16);
                const __bf16* pb = kbase + kk + bhi;
                ((v8bf*)&bb)[0] = *(const v8bf*)(pb);
                ((v8bf*)&bb)[1] = *(const v8bf*)(pb + 16);
                acc = __builtin_amdgcn_wmma_f32_16x16x32_bf16(
                    false, a, false, bb, (short)0, acc, false, false);
            }
        }
        int col = col0 + (lane & 15);
        bool kvalid; bool mk = true;
        if (isPad)       kvalid = false;
        else if (isVid)  kvalid = (col < HW_);
        else { kvalid = true; mk = tmask[b * LT_ + (col - 208)] != 0; }
        #pragma unroll
        for (int r = 0; r < 8; ++r) {
            int rr = r + ((lane & 16) ? 8 : 0);
            int qrow = q0 + rr;
            bool mq = true;
            if (textq) mq = (qrow < Lq) ? (tmask[b * LT_ + qrow] != 0) : true;
            float v = !kvalid ? -1e30f : ((mq && mk) ? acc[r] : -1e4f);
            S[rr * LKP + col] = v;
        }
    }
    __syncthreads();

    // ---- Phase 2: softmax (8 segs x 32 cols per row) ----
    int row = tid & 15, seg = tid >> 4;
    int cb = seg * 32;
    float mx = -3.4e38f;
    for (int c = 0; c < 32; ++c) mx = fmaxf(mx, S[row * LKP + cb + c]);
    red[row * 8 + seg] = mx;
    __syncthreads();
    if (tid < 16) {
        float m = red[tid * 8];
        #pragma unroll
        for (int i = 1; i < 8; ++i) m = fmaxf(m, red[tid * 8 + i]);
        rowstat[tid] = m;
    }
    __syncthreads();
    float m = rowstat[row];
    float sum = 0.f;
    for (int c = 0; c < 32; ++c) {
        float e = __expf(S[row * LKP + cb + c] - m);
        S[row * LKP + cb + c] = e;
        sum += e;
    }
    red[row * 8 + seg] = sum;
    __syncthreads();
    if (tid < 16) {
        float s = 0.f;
        #pragma unroll
        for (int i = 0; i < 8; ++i) s += red[tid * 8 + i];
        rowstat[tid] = SCALE_ / s;   // reference scales AFTER softmax
    }
    __syncthreads();
    float sc = rowstat[row];
    for (int c = 0; c < 32; ++c)
        P[row * LKP + cb + c] = f2bf(S[row * LKP + cb + c] * sc);
    __syncthreads();

    // ---- Phase 3: O = P * V^T  (wave -> dh tile of 16) ----
    int dh0 = wave * 16;
    v8f acc = (v8f){};
    const __bf16* vbase = VT + (size_t)(bv * D_ + h * DH_ + dh0 + (lane & 15)) * LKP;
    const __bf16* pbase = &P[(lane & 15) * LKP];
    for (int kk = 0; kk < LKP; kk += 32) {
        v16bf a, bb;
        const __bf16* pa = pbase + kk + alo;
        ((v8bf*)&a)[0] = *(const v8bf*)(pa);
        ((v8bf*)&a)[1] = *(const v8bf*)(pa + 16);
        const __bf16* pb = vbase + kk + bhi;
        ((v8bf*)&bb)[0] = *(const v8bf*)(pb);
        ((v8bf*)&bb)[1] = *(const v8bf*)(pb + 16);
        acc = __builtin_amdgcn_wmma_f32_16x16x32_bf16(
            false, a, false, bb, (short)0, acc, false, false);
    }
    #pragma unroll
    for (int r = 0; r < 8; ++r) {
        int qrow = q0 + r + ((lane & 16) ? 8 : 0);
        if (qrow < Lq)
            out[(size_t)(bv * Lq + qrow) * D_ + h * DH_ + dh0 + (lane & 15)] = acc[r];
    }
}

// ---------------------------------------------------------------------------
// Host launcher
// ---------------------------------------------------------------------------
extern "C" void kernel_launch(void* const* d_in, const int* in_sizes, int n_in,
                              void* d_out, int out_size, void* d_ws, size_t ws_size,
                              hipStream_t stream) {
    (void)in_sizes; (void)n_in; (void)out_size; (void)ws_size;
    const float* vid = (const float*)d_in[0];
    const float* txt = (const float*)d_in[1];
    const unsigned char* tmask = (const unsigned char*)d_in[3];
    const float* Wg[4] = { (const float*)d_in[4], (const float*)d_in[6],
                           (const float*)d_in[8], (const float*)d_in[10] };  // v2v,t2v,t2t,v2t
    const float* Bg[4] = { (const float*)d_in[5], (const float*)d_in[7],
                           (const float*)d_in[9], (const float*)d_in[11] };
    float* out = (float*)d_out;

    const size_t NV = (size_t)B_ * LV_ * HW_ * D_;   // 9,633,792
    const size_t NT = (size_t)B_ * LV_ * LT_ * D_;   // 1,572,864
    const size_t NW = (size_t)D_ * D_;               // 589,824
    const size_t NVT = (size_t)B_ * LV_ * D_ * LKP;  // 12,582,912

    char* p = (char*)d_ws;
    auto take = [&](size_t elems) { __bf16* r = (__bf16*)p; p += elems * 2; return r; };
    __bf16* vidb = take(NV);
    __bf16* txtb = take(NT);
    __bf16* Wb   = take(12 * NW);
    __bf16* Qvv  = take(NV);
    __bf16* Qvt  = take(NV);
    __bf16* Kvv  = take(NV);
    __bf16* Kvt  = take(NV);
    __bf16* Qtv  = take(NT);
    __bf16* Qtt  = take(NT);
    __bf16* Ktv  = take(NT);
    __bf16* Ktt  = take(NT);
    __bf16* vVT  = take(NVT);
    __bf16* tVT  = take(NVT);

    // ---- convert everything to bf16 ----
    cvt_kernel<<<2048, 256, 0, stream>>>(vid, vidb, (long long)NV);
    cvt_kernel<<<1024, 256, 0, stream>>>(txt, txtb, (long long)NT);
    for (int g = 0; g < 4; ++g)
        cvt_kernel<<<2048, 256, 0, stream>>>(Wg[g], Wb + (size_t)g * 3 * NW,
                                             (long long)(3 * NW));

    // ---- 12 projections ----
    auto wptr = [&](int g, int s) { return Wb + ((size_t)g * 3 + s) * NW; };
    ProjArgs pa;
    auto set = [&](int i, const __bf16* x, int T, int Ltok, int g, int s,
                   __bf16* o, int tr, int ko) {
        pa.d[i] = ProjDesc{ x, wptr(g, s), Bg[g] + s * D_, o, T, Ltok, tr, ko };
    };
    //    idx  x     T      Ltok  g  s  out   tr key_off
    set(0,  vidb, 12544, HW_, 0, 0, Qvv, 0, 0);    // v2v q on vid
    set(1,  vidb, 12544, HW_, 0, 1, Kvv, 0, 0);    // v2v k on vid
    set(2,  vidb, 12544, HW_, 0, 2, vVT, 1, 0);    // v2v v on vid  -> v_value vid part
    set(3,  vidb, 12544, HW_, 1, 0, Qvt, 0, 0);    // t2v q on vid
    set(4,  txtb, 2048,  LT_, 1, 1, Ktv, 0, 0);    // t2v k on txt
    set(5,  txtb, 2048,  LT_, 1, 2, vVT, 1, 208);  // t2v v on txt  -> v_value txt part
    set(6,  txtb, 2048,  LT_, 2, 0, Qtt, 0, 0);    // t2t q on txt
    set(7,  txtb, 2048,  LT_, 2, 1, Ktt, 0, 0);    // t2t k on txt
    set(8,  txtb, 2048,  LT_, 2, 2, tVT, 1, 208);  // t2t v on txt  -> t_value txt part
    set(9,  txtb, 2048,  LT_, 3, 0, Qtv, 0, 0);    // v2t q on txt
    set(10, vidb, 12544, HW_, 3, 1, Kvt, 0, 0);    // v2t k on vid
    set(11, vidb, 12544, HW_, 3, 2, tVT, 1, 0);    // v2t v on vid  -> t_value vid part

    proj_kernel<<<dim3(392, 3, 12), 256, 0, stream>>>(pa);

    // ---- attention ----
    // video queries: 13 q-tiles, 12 heads, 64 (b,v) pairs
    attn_kernel<<<dim3(13, H_, B_ * LV_), 128, 0, stream>>>(
        Qvv, Qvt, Kvv, Ktv, vVT, tmask, out, HW_, 0);
    // text queries: 2 q-tiles
    attn_kernel<<<dim3(2, H_, B_ * LV_), 128, 0, stream>>>(
        Qtv, Qtt, Kvt, Ktt, tVT, tmask, out + NV, LT_, 1);
}
